// GNN_68934225101285
// MI455X (gfx1250) — compile-verified, matrix operand order checked by reading, not run on hardware
//
#include <hip/hip_runtime.h>
#include <hip/hip_bf16.h>

// ---- problem constants (match reference) ----
#define NNODES 65536          // B * NPG
#define NPGR   256            // nodes per graph
#define NGRAPH 256            // graphs
#define DEG    16
#define NEDGE  (NNODES * DEG) // 1,048,576
#define DIM    128
#define KKEEP  128            // TopK ratio 0.5
#define NPOOL  (NGRAPH * KKEEP) // 32768
#define OUTD   12
#define BNEPS  1e-5f

typedef __attribute__((ext_vector_type(16))) _Float16 v16h;
typedef __attribute__((ext_vector_type(8)))  _Float16 v8h;
typedef __attribute__((ext_vector_type(8)))  float    v8f;

// ------------------------------------------------------------------
__global__ void zero_f32(float* __restrict__ p, int n) {
  int i = blockIdx.x * blockDim.x + threadIdx.x;
  if (i < n) p[i] = 0.0f;
}

// AtomEncoder: h[n][d] = sum_f atom_emb[f][x[n][f]][d]
__global__ __launch_bounds__(DIM)
void atom_encoder(const int* __restrict__ x, const float* __restrict__ emb,
                  float* __restrict__ h) {
  const int n = blockIdx.x;
  const int d = threadIdx.x;
  float s = 0.0f;
#pragma unroll
  for (int f = 0; f < 9; ++f) {
    const int v = x[n * 9 + f];
    s += emb[(f * 128 + v) * DIM + d];
  }
  h[(long long)n * DIM + d] = s;
}

// scatter-add aggregation: agg[dst] += h[src] (optionally through inv remap)
__global__ void aggregate(const float* __restrict__ h,
                          const int* __restrict__ src, const int* __restrict__ dst,
                          const int* __restrict__ inv, float* __restrict__ agg,
                          int nOut, long long nWork) {
  long long gid = (long long)blockIdx.x * blockDim.x + threadIdx.x;
  if (gid >= nWork) return;
  int e = (int)(gid >> 5);        // DIM/4 = 32 channel-quads per edge
  int c = (int)(gid & 31) << 2;
  int s = src[e], d = dst[e];
  if (inv) {
    s = inv[s]; d = inv[d];
    if (s >= nOut || d >= nOut) return;   // dropped edge -> dummy segment
  }
  const float4 v = *(const float4*)(h + (long long)s * DIM + c);
  float* a = agg + (long long)d * DIM + c;
  unsafeAtomicAdd(a + 0, v.x);
  unsafeAtomicAdd(a + 1, v.y);
  unsafeAtomicAdd(a + 2, v.z);
  unsafeAtomicAdd(a + 3, v.w);
}

// z = ((1+eps)*h + agg) @ W + b  via v_wmma_f32_16x16x32_f16;
// also accumulates per-channel sum(z), sum(z^2) into `sums` (for BatchNorm).
__global__ __launch_bounds__(256)
void gin_gemm(const float* __restrict__ h, const float* __restrict__ agg,
              const float* __restrict__ W, const float* __restrict__ bias,
              const float* __restrict__ epsArr, int layer,
              float* __restrict__ z, float* __restrict__ sums) {
  __shared__ __align__(16) _Float16 smem[2 * 128 * 128];  // 64 KB
  _Float16* Ah = smem;              // A tile: [row][k], row-major, f16
  _Float16* Bt = smem + 128 * 128;  // W^T:    [n][k],  row-major, f16

  const int tid = threadIdx.x;
  const long long rowBase = (long long)blockIdx.x * 128;
  const float epv = 1.0f + epsArr[layer];
  const float* Wl = W + (long long)layer * DIM * DIM;

  // stage A = (1+eps)*h + agg  and  Bt[n][k] = W[k][n]  (coalesced global reads)
  for (int idx = tid; idx < 128 * 128; idx += 256) {
    const int r = idx >> 7, c = idx & 127;
    const long long gi = (rowBase + r) * DIM + c;
    Ah[r * 128 + c] = (_Float16)(epv * h[gi] + agg[gi]);
    const int k = r, n = c;                 // read Wl row-major
    Bt[n * 128 + k] = (_Float16)Wl[idx];
  }
  __syncthreads();

  const int wave = tid >> 5;
  const int lane = tid & 31;
  const int g    = lane >> 4;     // lane group (K-run selector)
  const int m    = wave * 16 + (lane & 15);

  float ps1[8], ps2[8];
#pragma unroll
  for (int i = 0; i < 8; ++i) { ps1[i] = 0.0f; ps2[i] = 0.0f; }

  for (int ct = 0; ct < 8; ++ct) {
    const int n = ct * 16 + (lane & 15);
    v8f acc = {};
#pragma unroll
    for (int kk = 0; kk < 128; kk += 32) {
      // A fragment: lane-group g holds K in [kk+8g, kk+8g+8) and [kk+16+8g, +8)
      const v8h a0 = *(const v8h*)(Ah + m * 128 + kk + 8 * g);
      const v8h a1 = *(const v8h*)(Ah + m * 128 + kk + 16 + 8 * g);
      const v8h b0 = *(const v8h*)(Bt + n * 128 + kk + 8 * g);
      const v8h b1 = *(const v8h*)(Bt + n * 128 + kk + 16 + 8 * g);
      const v16h av = __builtin_shufflevector(a0, a1, 0,1,2,3,4,5,6,7,8,9,10,11,12,13,14,15);
      const v16h bv = __builtin_shufflevector(b0, b1, 0,1,2,3,4,5,6,7,8,9,10,11,12,13,14,15);
      acc = __builtin_amdgcn_wmma_f32_16x16x32_f16(false, av, false, bv,
                                                   (short)0, acc, false, false);
    }
    const float bb = bias[layer * DIM + n];
#pragma unroll
    for (int i = 0; i < 8; ++i) {
      const float val = acc[i] + bb;
      const long long row = rowBase + wave * 16 + i + 8 * g;  // C layout: VGPR i -> M=i+8g
      z[row * DIM + n] = val;
      ps1[ct] += val;
      ps2[ct] += val * val;
    }
  }

  // block reduction of per-channel stats (reuse A region of LDS)
  __syncthreads();
  float* red1 = (float*)smem;          // [256][8]
  float* red2 = red1 + 256 * 8;        // [256][8]
#pragma unroll
  for (int ct = 0; ct < 8; ++ct) {
    red1[tid * 8 + ct] = ps1[ct];
    red2[tid * 8 + ct] = ps2[ct];
  }
  __syncthreads();
  if (tid < DIM) {
    const int c16 = tid & 15, ct = tid >> 4;   // channel = ct*16 + c16 = tid
    float s1 = 0.0f, s2 = 0.0f;
#pragma unroll
    for (int w8 = 0; w8 < 8; ++w8)
#pragma unroll
      for (int gg = 0; gg < 2; ++gg) {
        const int l = w8 * 32 + gg * 16 + c16;
        s1 += red1[l * 8 + ct];
        s2 += red2[l * 8 + ct];
      }
    unsafeAtomicAdd(&sums[tid], s1);
    unsafeAtomicAdd(&sums[DIM + tid], s2);
  }
}

// BatchNorm (training, biased var) + ReLU:  h = relu((z-mu)*rsqrt(var+eps)*g + b)
__global__ void bn_relu(const float* __restrict__ z, const float* __restrict__ sums,
                        const float* __restrict__ gamma, const float* __restrict__ beta,
                        int layer, float* __restrict__ h, int M) {
  const long long gid = (long long)blockIdx.x * blockDim.x + threadIdx.x;
  const int c = (int)(gid & 127);
  const float invM = 1.0f / (float)M;
  const float mu  = sums[c] * invM;
  const float var = sums[DIM + c] * invM - mu * mu;
  const float sc  = rsqrtf(var + BNEPS) * gamma[layer * DIM + c];
  const float val = (z[gid] - mu) * sc + beta[layer * DIM + c];
  h[gid] = fmaxf(val, 0.0f);
}

// TopKPooling ratio 0.5: score = h.w/||w||; keep top K per graph (stable rank),
// gate by tanh(score); build inverse permutation for edge remap.
__global__ __launch_bounds__(NPGR)
void topk_pool(const float* __restrict__ h, const float* __restrict__ w,
               float* __restrict__ h2, int* __restrict__ inv) {
  __shared__ float sw[DIM];
  __shared__ float ssc[NPGR];
  const int gidx = blockIdx.x;
  const int t = threadIdx.x;
  if (t < DIM) sw[t] = w[t];
  __syncthreads();
  float nsq = 0.0f;
#pragma unroll 8
  for (int d = 0; d < DIM; ++d) nsq += sw[d] * sw[d];
  const float invn = rsqrtf(nsq);
  const int node = gidx * NPGR + t;
  float dot = 0.0f;
#pragma unroll 8
  for (int d = 0; d < DIM; ++d) dot += h[(long long)node * DIM + d] * sw[d];
  const float sc = dot * invn;
  ssc[t] = sc;
  __syncthreads();
  int rank = 0;
  for (int j = 0; j < NPGR; ++j) {
    const float o = ssc[j];
    rank += (o > sc) || (o == sc && j < t);   // stable top_k tie-break
  }
  if (rank < KKEEP) {
    const int ni = gidx * KKEEP + rank;
    inv[node] = ni;
    const float gate = tanhf(sc);
#pragma unroll 4
    for (int d = 0; d < DIM; ++d)
      h2[(long long)ni * DIM + d] = h[(long long)node * DIM + d] * gate;
  } else {
    inv[node] = NPOOL;   // sentinel: dropped node
  }
}

// global_mean_pool (over K kept nodes) -> @ out_W + out_b -> sigmoid
__global__ __launch_bounds__(DIM)
void readout(const float* __restrict__ h2, const float* __restrict__ oW,
             const float* __restrict__ ob, float* __restrict__ out) {
  __shared__ float pooled[DIM];
  const int g = blockIdx.x, d = threadIdx.x;
  float s = 0.0f;
  for (int r = 0; r < KKEEP; ++r)
    s += h2[((long long)g * KKEEP + r) * DIM + d];
  pooled[d] = s * (1.0f / (float)KKEEP);
  __syncthreads();
  if (d < OUTD) {
    float acc = ob[d];
#pragma unroll 8
    for (int k = 0; k < DIM; ++k) acc += pooled[k] * oW[k * OUTD + d];
    out[g * OUTD + d] = 1.0f / (1.0f + expf(-acc));
  }
}

// ------------------------------------------------------------------
extern "C" void kernel_launch(void* const* d_in, const int* in_sizes, int n_in,
                              void* d_out, int out_size, void* d_ws, size_t ws_size,
                              hipStream_t stream) {
  (void)in_sizes; (void)n_in; (void)out_size; (void)ws_size;
  // setup_inputs() dict order:
  const int*   x        = (const int*)d_in[0];     // [N,9]
  const int*   edge     = (const int*)d_in[1];     // [2,E]
  // d_in[2] edge_attr, d_in[3] batch: unused (DCE'd / derivable)
  const float* atom_emb = (const float*)d_in[4];   // [9,128,D]
  // d_in[5] bond_emb: unused
  const float* convW    = (const float*)d_in[6];   // [5,D,D]
  const float* convB    = (const float*)d_in[7];   // [5,D]
  const float* gamma    = (const float*)d_in[8];   // [5,D]
  const float* beta     = (const float*)d_in[9];   // [5,D]
  const float* epsArr   = (const float*)d_in[10];  // [5]
  const float* topkw    = (const float*)d_in[11];  // [D]
  const float* outW     = (const float*)d_in[12];  // [D,OUT]
  const float* outB     = (const float*)d_in[13];  // [OUT]
  float* out = (float*)d_out;

  // workspace layout (needs ~67.4 MB)
  char* ws = (char*)d_ws;
  float* P     = (float*)ws;                                        // 32 MB (h)
  float* Q     = (float*)(ws + (size_t)NNODES * DIM * 4);           // 32 MB (agg/z)
  int*   inv   = (int*)  (ws + (size_t)2 * NNODES * DIM * 4);       // 256 KB
  float* stats = (float*)(ws + (size_t)2 * NNODES * DIM * 4 + (size_t)NNODES * 4);

  const int* src = edge;
  const int* dst = edge + NEDGE;

  atom_encoder<<<NNODES, DIM, 0, stream>>>(x, atom_emb, P);

  const long long nwFull = (long long)NEDGE * 32;
  const int aggBlocks = (int)((nwFull + 255) / 256);

  // ---- 3 GIN layers on the full graph: h in P, agg/z in Q ----
  for (int layer = 0; layer < 3; ++layer) {
    zero_f32<<<(NNODES * DIM + 255) / 256, 256, 0, stream>>>(Q, NNODES * DIM);
    zero_f32<<<1, 256, 0, stream>>>(stats, 2 * DIM);
    aggregate<<<aggBlocks, 256, 0, stream>>>(P, src, dst, nullptr, Q, NNODES, nwFull);
    gin_gemm<<<NNODES / 128, 256, 0, stream>>>(P, Q, convW, convB, epsArr, layer, Q, stats);
    bn_relu<<<NNODES * DIM / 256, 256, 0, stream>>>(Q, stats, gamma, beta, layer, P, NNODES);
  }

  // ---- TopK pooling: read P, write gated h2 into Q, build inv ----
  topk_pool<<<NGRAPH, NPGR, 0, stream>>>(P, topkw, Q, inv);

  // ---- 2 GIN layers on pooled graph: h in Q, agg/z in P ----
  for (int layer = 3; layer < 5; ++layer) {
    zero_f32<<<(NPOOL * DIM + 255) / 256, 256, 0, stream>>>(P, NPOOL * DIM);
    zero_f32<<<1, 256, 0, stream>>>(stats, 2 * DIM);
    aggregate<<<aggBlocks, 256, 0, stream>>>(Q, src, dst, inv, P, NPOOL, nwFull);
    gin_gemm<<<NPOOL / 128, 256, 0, stream>>>(Q, P, convW, convB, epsArr, layer, P, stats);
    bn_relu<<<NPOOL * DIM / 256, 256, 0, stream>>>(P, stats, gamma, beta, layer, Q, NPOOL);
  }

  readout<<<NGRAPH, DIM, 0, stream>>>(Q, outW, outB, out);
}
